// SpatialBottleneck_65000035057955
// MI455X (gfx1250) — compile-verified, matrix operand order checked by reading, not run on hardware
//
#include <hip/hip_runtime.h>

// ---------- vector types ----------
typedef __attribute__((ext_vector_type(16))) _Float16 v16h;
typedef __attribute__((ext_vector_type(4)))  _Float16 v4h;
typedef __attribute__((ext_vector_type(8)))  float    v8f;
typedef __attribute__((ext_vector_type(4)))  float    f32x4;
typedef __attribute__((ext_vector_type(4)))  unsigned int u32x4;
typedef __attribute__((ext_vector_type(4)))  int      i32x4;
typedef __attribute__((address_space(1)))    i32x4    glb_i32x4;
typedef __attribute__((address_space(3)))    i32x4    lds_i32x4;

// ---------- CDNA5 async global->LDS copy (ASYNCcnt path) ----------
#define HAVE_ASYNC_LDS __has_builtin(__builtin_amdgcn_global_load_async_to_lds_b128)

__device__ __forceinline__ void cp16_g2l(const _Float16* g, _Float16* l) {
#if HAVE_ASYNC_LDS
  __builtin_amdgcn_global_load_async_to_lds_b128(
      (glb_i32x4*)g,   // global src (AS1 int4*)
      (lds_i32x4*)l,   // LDS dst   (AS3 int4*)
      0, 0);
#else
  *(u32x4*)l = *(const u32x4*)g;   // synchronous fallback
#endif
}

__device__ __forceinline__ void wait_async_lds() {
#if HAVE_ASYNC_LDS
#if __has_builtin(__builtin_amdgcn_s_wait_asynccnt)
  __builtin_amdgcn_s_wait_asynccnt(0);
#else
  asm volatile("s_wait_asynccnt 0x0" ::: "memory");
#endif
#endif
}

// ---------- problem constants ----------
// N=32 H=32 W=32 Cin=1024 Cb=256 ; M = N*H*W = 32768
constexpr int M_PIX = 32768;
constexpr int BM = 128, BN = 64, KC = 32;
constexpr int SA = 40;                 // padded LDS row stride in halfs (80B, bank-conflict-free)
constexpr int A_HALFS  = BM * SA;      // 5120
constexpr int B_HALFS  = BN * SA;      // 2560
constexpr int BUF_HALFS = A_HALFS + B_HALFS;  // 7680 halfs = 15360 B per buffer

// ---------- prep kernels ----------
__global__ __launch_bounds__(256) void cvt_x_kernel(const float* __restrict__ x,
                                                    _Float16* __restrict__ xh, int n4) {
  int i = blockIdx.x * blockDim.x + threadIdx.x;
  if (i < n4) {
    f32x4 f = ((const f32x4*)x)[i];
    v4h o;
    o[0] = (_Float16)f[0]; o[1] = (_Float16)f[1];
    o[2] = (_Float16)f[2]; o[3] = (_Float16)f[3];
    ((v4h*)xh)[i] = o;
  }
}

// transpose HWIO weights to [N][K] f16 (K contiguous per output channel)
__global__ __launch_bounds__(256) void prep_w_kernel(const float* __restrict__ w1,
                                                     const float* __restrict__ w2,
                                                     const float* __restrict__ w3,
                                                     _Float16* __restrict__ w1t,
                                                     _Float16* __restrict__ w2t,
                                                     _Float16* __restrict__ w3t,
                                                     _Float16* __restrict__ zpad) {
  int i = blockIdx.x * blockDim.x + threadIdx.x;
  if (i < 256 * 1024) {                       // w1: (1,1,1024,256) -> w1t[co][ci]
    int co = i >> 10, ci = i & 1023;
    w1t[i] = (_Float16)w1[ci * 256 + co];
  }
  if (i < 256 * 2304) {                       // w2: (3,3,256,256) -> w2t[co][tap][ci]
    int co = i / 2304, rr = i % 2304;
    int tap = rr >> 8, ci = rr & 255;
    w2t[i] = (_Float16)w2[(tap * 256 + ci) * 256 + co];
  }
  if (i < 1024 * 256) {                       // w3: (1,1,256,1024) -> w3t[co][ci]
    int co = i >> 8, ci = i & 255;
    w3t[i] = (_Float16)w3[ci * 1024 + co];
  }
  if (i < 128) zpad[i] = (_Float16)0.0f;      // 256B zero pad for conv2 halo
}

// ---------- fused conv-as-GEMM kernel ----------
// MODE 0: conv1x1 1024->256 (A=xh)        out = relu(acc*s+b) -> o1 (f16)
// MODE 1: conv3x3  256->256 (A=o1 gather) out = relu(acc*s+b) -> o2 (f16)
// MODE 2: conv1x1  256->1024 (A=o2)       out = relu(acc*s+b+x) -> d_out (f32)
template <int MODE>
__global__ __launch_bounds__(256) void conv_gemm(
    const _Float16* __restrict__ A, const _Float16* __restrict__ Bt,
    const float* __restrict__ scale, const float* __restrict__ bias,
    const _Float16* __restrict__ resid, const _Float16* __restrict__ zpad,
    _Float16* __restrict__ outh, float* __restrict__ outf) {
  constexpr int KA   = (MODE == 0) ? 1024 : 256;                  // A row stride (halfs)
  constexpr int KB   = (MODE == 0) ? 1024 : (MODE == 1 ? 2304 : 256);
  constexpr int NOUT = (MODE == 2) ? 1024 : 256;
  constexpr int NK   = (MODE == 0) ? 32 : (MODE == 1 ? 72 : 8);   // K chunks of 32

  __shared__ __align__(16) _Float16 smem[2 * BUF_HALFS];

  const int tid  = threadIdx.x;
  const int lane = tid & 31;
  const int wid  = tid >> 5;
  const int wm   = wid & 3;    // 4 wave rows  (32 M each)
  const int wn   = wid >> 2;   // 2 wave cols  (32 N each)
  const int h    = lane >> 4;  // half-wave
  const int r    = lane & 15;

  const int m0 = blockIdx.x * BM;
  const int n0 = blockIdx.y * BN;

  v8f acc[2][2] = {};

  auto issue = [&](int kk, int buf) {
    _Float16* base = smem + buf * BUF_HALFS;
    {  // B tile: 64 rows x 32 halfs = 256 b128 segments, 1 per thread
      int seg = tid, row = seg >> 2, c8 = seg & 3;
      const _Float16* g = Bt + (size_t)(n0 + row) * KB + kk * KC + c8 * 8;
      cp16_g2l(g, base + A_HALFS + row * SA + c8 * 8);
    }
#pragma unroll
    for (int i = 0; i < 2; ++i) {  // A tile: 128 rows x 32 halfs = 512 segments
      int seg = tid + 256 * i, row = seg >> 2, c8 = seg & 3;
      const _Float16* g;
      if constexpr (MODE == 1) {
        int tap = kk >> 3, kc = kk & 7;
        int dy = tap / 3 - 1, dx = tap % 3 - 1;
        int p  = m0 + row;
        int hh = ((p >> 5) & 31) + dy, ww = (p & 31) + dx;
        if ((unsigned)hh < 32u && (unsigned)ww < 32u) {
          int q = (p & ~1023) + (hh << 5) + ww;   // shifted pixel index
          g = A + (size_t)q * 256 + kc * 32 + c8 * 8;
        } else {
          g = zpad + c8 * 8;                      // zero padding (SAME)
        }
      } else {
        g = A + (size_t)(m0 + row) * KA + kk * KC + c8 * 8;
      }
      cp16_g2l(g, base + row * SA + c8 * 8);
    }
  };

  issue(0, 0);
  for (int kk = 0; kk < NK; ++kk) {
    wait_async_lds();
    __syncthreads();
    if (kk + 1 < NK) issue(kk + 1, (kk + 1) & 1);   // prefetch next chunk

    const _Float16* base = smem + (kk & 1) * BUF_HALFS;
    v16h afrag[2], bfrag[2];
#pragma unroll
    for (int ms = 0; ms < 2; ++ms) {   // A frag: lane half h holds K {8h..8h+7} U {16+8h..}
      const _Float16* ap = base + (wm * 32 + ms * 16 + r) * SA + 8 * h;
      union { u32x4 u[2]; v16h v; } u;
      u.u[0] = *(const u32x4*)ap;
      u.u[1] = *(const u32x4*)(ap + 16);
      afrag[ms] = u.v;
    }
#pragma unroll
    for (int ns = 0; ns < 2; ++ns) {
      const _Float16* bp = base + A_HALFS + (wn * 32 + ns * 16 + r) * SA + 8 * h;
      union { u32x4 u[2]; v16h v; } u;
      u.u[0] = *(const u32x4*)bp;
      u.u[1] = *(const u32x4*)(bp + 16);
      bfrag[ns] = u.v;
    }
#pragma unroll
    for (int ms = 0; ms < 2; ++ms)
#pragma unroll
      for (int ns = 0; ns < 2; ++ns)
        acc[ms][ns] = __builtin_amdgcn_wmma_f32_16x16x32_f16(
            false, afrag[ms], false, bfrag[ns], (short)0, acc[ms][ns], false, false);
  }

  // epilogue: C layout -> VGPR e: lanes0-15 M=e, lanes16-31 M=e+8 ; N = lane&15
#pragma unroll
  for (int ns = 0; ns < 2; ++ns) {
    int n = n0 + wn * 32 + ns * 16 + r;
    float s = scale[n], b = bias[n];
#pragma unroll
    for (int ms = 0; ms < 2; ++ms) {
      int mb = m0 + wm * 32 + ms * 16 + 8 * h;
#pragma unroll
      for (int e = 0; e < 8; ++e) {
        int m = mb + e;
        float v = fmaf(acc[ms][ns][e], s, b);
        if constexpr (MODE == 2) {
          v += (float)resid[(size_t)m * 1024 + n];
          outf[(size_t)m * NOUT + n] = v > 0.0f ? v : 0.0f;
        } else {
          outh[(size_t)m * NOUT + n] = (_Float16)(v > 0.0f ? v : 0.0f);
        }
      }
    }
  }
}

// ---------- host launcher ----------
extern "C" void kernel_launch(void* const* d_in, const int* in_sizes, int n_in,
                              void* d_out, int out_size, void* d_ws, size_t ws_size,
                              hipStream_t stream) {
  const float* x  = (const float*)d_in[0];
  const float* w1 = (const float*)d_in[1];
  const float* w2 = (const float*)d_in[2];
  const float* w3 = (const float*)d_in[3];
  const float* s1 = (const float*)d_in[4];
  const float* b1 = (const float*)d_in[5];
  const float* s2 = (const float*)d_in[6];
  const float* b2 = (const float*)d_in[7];
  const float* s3 = (const float*)d_in[8];
  const float* b3 = (const float*)d_in[9];

  char* ws = (char*)d_ws;
  size_t off = 0;
  _Float16* xh  = (_Float16*)(ws + off); off += (size_t)M_PIX * 1024 * 2;  // 64 MB
  _Float16* o1  = (_Float16*)(ws + off); off += (size_t)M_PIX * 256 * 2;   // 16 MB
  _Float16* o2  = (_Float16*)(ws + off); off += (size_t)M_PIX * 256 * 2;   // 16 MB
  _Float16* w1t = (_Float16*)(ws + off); off += (size_t)256 * 1024 * 2;
  _Float16* w2t = (_Float16*)(ws + off); off += (size_t)256 * 2304 * 2;
  _Float16* w3t = (_Float16*)(ws + off); off += (size_t)1024 * 256 * 2;
  _Float16* zp  = (_Float16*)(ws + off); off += 256;

  cvt_x_kernel<<<32768, 256, 0, stream>>>(x, xh, M_PIX * 1024 / 4);
  prep_w_kernel<<<(256 * 2304 + 255) / 256, 256, 0, stream>>>(w1, w2, w3, w1t, w2t, w3t, zp);

  conv_gemm<0><<<dim3(M_PIX / BM, 256 / BN), 256, 0, stream>>>(
      xh, w1t, s1, b1, nullptr, zp, o1, nullptr);
  conv_gemm<1><<<dim3(M_PIX / BM, 256 / BN), 256, 0, stream>>>(
      o1, w2t, s2, b2, nullptr, zp, o2, nullptr);
  conv_gemm<2><<<dim3(M_PIX / BM, 1024 / BN), 256, 0, stream>>>(
      o2, w3t, s3, b3, xh, zp, nullptr, (float*)d_out);
}
